// ConvNDAttention_26766236189066
// MI455X (gfx1250) — compile-verified, matrix operand order checked by reading, not run on hardware
//
#include <hip/hip_runtime.h>

// Local 7x7 window attention, X:[4,64,64,256] f32 -> O:[4,58,58,256] f32.
// Dense 16-query banded-mask attention tiles on v_wmma_f32_16x16x32_bf16.
// f32 -> bf16 operand conversion via packed hardware cvt (v_cvt_pk_bf16_f32).

typedef __attribute__((ext_vector_type(16))) __bf16 v16bf;
typedef __attribute__((ext_vector_type(2)))  __bf16 v2bf;
typedef __attribute__((ext_vector_type(8)))  float  v8f;

#define BATCH 4
#define Hh 64
#define Ww 64
#define Cc 256
#define HP 58
#define WP 58

union V16 { v16bf v; v2bf p[8]; unsigned u[8]; };

__device__ __forceinline__ int imin(int a, int b) { return a < b ? a : b; }

// two f32 -> packed bf16 pair (one VALU op on CDNA5)
__device__ __forceinline__ v2bf cvt2(float lo, float hi) {
#if __has_builtin(__builtin_amdgcn_cvt_pk_bf16_f32)
    return __builtin_amdgcn_cvt_pk_bf16_f32(lo, hi);
#else
    v2bf r; r[0] = (__bf16)lo; r[1] = (__bf16)hi; return r;
#endif
}

// Build one lane's 16-bf16 WMMA operand from 32 contiguous f32 "K" values at p.
// Per ISA layout (16-bit A 16x32 / B 32x16), lane half h holds K groups
// {8h..8h+7} in uint slots 0..3 and {16+8h..16+8h+7} in slots 4..7, 2 per slot.
__device__ __forceinline__ v16bf load_op32(const float* __restrict__ p, int h) {
    V16 r;
    const float* a = p + 8 * h;
    const float* b = p + 16 + 8 * h;
#pragma unroll
    for (int v = 0; v < 4; ++v) {
        r.p[v]     = cvt2(a[2 * v], a[2 * v + 1]);
        r.p[v + 4] = cvt2(b[2 * v], b[2 * v + 1]);
    }
    return r.v;
}

__global__ __launch_bounds__(256)
void local_attn_wmma(const float* __restrict__ X, float* __restrict__ O) {
    const int j0   = blockIdx.x * 16;   // first query col of this tile
    const int iq   = blockIdx.y;        // query row (0..57)
    const int bb   = blockIdx.z;
    const int tid  = threadIdx.x;
    const int wave = tid >> 5;          // 0..7
    const int lane = tid & 31;
    const int h    = lane >> 4;         // K-half selector
    const int l15  = lane & 15;         // M (A) or N (B/D) index

    __shared__ float    s_s[16][7][32];   // scores  [query][keyrow][keycol] 14KB
    __shared__ unsigned s_w[16][7][16];   // bf16x2 softmax weights           7KB

    const float* Xb = X + (size_t)bb * Hh * Ww * Cc;

    // ---------------- Phase 1: S[m][r][n] = (Q_m . K_{r,n}) / 16 ------------
    // 14 score tiles t = r*2 + nt ; wave handles t=wave and t=wave+8.
    const int t0 = wave, t1 = wave + 8;
    v8f acc0 = {}; v8f acc1 = {};

    const int qj = imin(j0 + l15, WP - 1) + 3;                    // clamp pad queries
    const float* qp = Xb + ((size_t)(iq + 3) * Ww + qj) * Cc;     // Q row for M=l15

    const int r0 = t0 >> 1, nt0 = t0 & 1;
    const int r1 = t1 >> 1, nt1 = t1 & 1;
    const int kj0 = imin(j0 + nt0 * 16 + l15, Ww - 1);
    const int kj1 = imin(j0 + nt1 * 16 + l15, Ww - 1);
    const float* kp0 = Xb + ((size_t)(iq + r0) * Ww + kj0) * Cc;
    const float* kp1 = Xb + ((size_t)(iq + r1) * Ww + kj1) * Cc;

#pragma unroll
    for (int kc = 0; kc < 8; ++kc) {                              // 256 ch / 32
        v16bf a  = load_op32(qp + kc * 32, h);
        v16bf b0 = load_op32(kp0 + kc * 32, h);
        acc0 = __builtin_amdgcn_wmma_f32_16x16x32_bf16(
                   false, a, false, b0, (short)0, acc0, false, false);
        if (t1 < 14) {                                            // wave-uniform
            v16bf b1 = load_op32(kp1 + kc * 32, h);
            acc1 = __builtin_amdgcn_wmma_f32_16x16x32_bf16(
                       false, a, false, b1, (short)0, acc1, false, false);
        }
    }

    // mask + store scores (D layout: VGPR v -> M=v+8h, lane&15 -> N)
#pragma unroll
    for (int v = 0; v < 8; ++v) {
        int m = v + 8 * h;
        int n = nt0 * 16 + l15;
        int d = n - m;                                            // key offset in window
        bool ok = (d >= 0) & (d <= 6) & !((r0 == 3) & (d == 3)) & ((j0 + m) < WP);
        s_s[m][r0][n] = ok ? acc0[v] * 0.0625f : -1e30f;
    }
    if (t1 < 14) {
#pragma unroll
        for (int v = 0; v < 8; ++v) {
            int m = v + 8 * h;
            int n = nt1 * 16 + l15;
            int d = n - m;
            bool ok = (d >= 0) & (d <= 6) & !((r1 == 3) & (d == 3)) & ((j0 + m) < WP);
            s_s[m][r1][n] = ok ? acc1[v] * 0.0625f : -1e30f;
        }
    }
    __syncthreads();

    // ---------------- Phase 2: softmax over the 7*32 masked keys ------------
    {
        int m = tid >> 4;                                         // query 0..15
        int t = tid & 15;                                         // 16 lanes/query
        float sv[14];
#pragma unroll
        for (int r = 0; r < 7; ++r) {
            sv[2 * r]     = s_s[m][r][2 * t];
            sv[2 * r + 1] = s_s[m][r][2 * t + 1];
        }
        float mx = -1e30f;
#pragma unroll
        for (int k = 0; k < 14; ++k) mx = fmaxf(mx, sv[k]);
#pragma unroll
        for (int o = 8; o >= 1; o >>= 1) mx = fmaxf(mx, __shfl_xor(mx, o, 16));
        float ev[14];
        float sum = 0.f;
#pragma unroll
        for (int k = 0; k < 14; ++k) { ev[k] = __expf(sv[k] - mx); sum += ev[k]; }
#pragma unroll
        for (int o = 8; o >= 1; o >>= 1) sum += __shfl_xor(sum, o, 16);
        float inv = 1.0f / sum;
#pragma unroll
        for (int r = 0; r < 7; ++r)
            s_w[m][r][t] = __builtin_bit_cast(
                unsigned, cvt2(ev[2 * r] * inv, ev[2 * r + 1] * inv));
    }
    __syncthreads();

    // ---------------- Phase 3: O = W . V  (keys double as values) -----------
    // Each wave owns 2 channel N-tiles; accumulate K=32 keys over 7 rows.
#pragma unroll
    for (int q = 0; q < 2; ++q) {
        const int c0 = (wave * 2 + q) * 16;
        v8f acc = {};
#pragma unroll
        for (int row = 0; row < 7; ++row) {
            // A = weights: M=l15 (query), uint slot v holds K pair p
            V16 aw;
#pragma unroll
            for (int v = 0; v < 4; ++v) {
                aw.u[v]     = s_w[l15][row][4 * h + v];
                aw.u[v + 4] = s_w[l15][row][8 + 4 * h + v];
            }
            // B = values: N=l15 (channel), K = key index; coalesced dword/lane
            const float* vrow = Xb + (size_t)(iq + row) * Ww * Cc + c0 + l15;
            V16 bw;
#pragma unroll
            for (int v = 0; v < 4; ++v) {
                int ka = 8 * h + 2 * v;
                int kb = 16 + 8 * h + 2 * v;
                bw.p[v]     = cvt2(vrow[(size_t)imin(j0 + ka,     Ww - 1) * Cc],
                                   vrow[(size_t)imin(j0 + ka + 1, Ww - 1) * Cc]);
                bw.p[v + 4] = cvt2(vrow[(size_t)imin(j0 + kb,     Ww - 1) * Cc],
                                   vrow[(size_t)imin(j0 + kb + 1, Ww - 1) * Cc]);
            }
            acc = __builtin_amdgcn_wmma_f32_16x16x32_bf16(
                      false, aw.v, false, bw.v, (short)0, acc, false, false);
        }
        // store D: M = v+8h (query), N = l15 (channel)
#pragma unroll
        for (int v = 0; v < 8; ++v) {
            int m = v + 8 * h;
            if (j0 + m < WP) {
                O[(((size_t)bb * HP + iq) * WP + (j0 + m)) * Cc + c0 + l15] = acc[v];
            }
        }
    }
}

extern "C" void kernel_launch(void* const* d_in, const int* in_sizes, int n_in,
                              void* d_out, int out_size, void* d_ws, size_t ws_size,
                              hipStream_t stream) {
    const float* X = (const float*)d_in[0];
    float* O = (float*)d_out;
    (void)in_sizes; (void)n_in; (void)out_size; (void)d_ws; (void)ws_size;
    dim3 grid((WP + 15) / 16, HP, BATCH);   // 4 x 58 x 4 workgroups
    local_attn_wmma<<<grid, 256, 0, stream>>>(X, O);
}
